// ViTBlock_30064771072387
// MI455X (gfx1250) — matrix operand order repaired and checked by reading, not compiled, not run on hardware
//
#include <hip/hip_runtime.h>
#include <math.h>

// ---- problem constants (from reference) ----
#define CDIM   768
#define TDIM   2048
#define BDIM   4
#define HDIM   12
#define DHDIM  64
#define MROWS  (BDIM * TDIM)      // 8192
#define QKVLD  (3 * CDIM)         // 2304
#define FFDIM  (4 * CDIM)         // 3072

typedef __attribute__((ext_vector_type(16))) __bf16 v16bf;
typedef __attribute__((ext_vector_type(8)))  float  v8f;

union Frag {
  v16bf v;
  uint4 q[2];
};

static __device__ __forceinline__ unsigned short f32_bf16(float f) {
  unsigned int u = __float_as_uint(f);
  u += 0x7FFFu + ((u >> 16) & 1u);          // round-to-nearest-even
  return (unsigned short)(u >> 16);
}

// ---------------- async global -> LDS copy (CDNA5, ASYNCcnt) ----------------
#if defined(__has_builtin)
#if __has_builtin(__builtin_amdgcn_global_load_async_to_lds_b128)
#define USE_ASYNC_BUILTIN 1
#endif
#endif

typedef int v4i __attribute__((__vector_size__(4 * sizeof(int))));
typedef __attribute__((address_space(1))) v4i* gv4i_ptr;
typedef __attribute__((address_space(3))) v4i* lv4i_ptr;
typedef __attribute__((address_space(3))) unsigned short LASU16;

static __device__ __forceinline__ void cp_async16(const unsigned short* g,
                                                  unsigned short* l) {
#if defined(USE_ASYNC_BUILTIN)
  __builtin_amdgcn_global_load_async_to_lds_b128((gv4i_ptr)(void*)(unsigned short*)g,
                                                 (lv4i_ptr)(void*)l, 0, 0);
#else
  unsigned lo = (unsigned)(unsigned long long)(LASU16*)l;
  asm volatile("global_load_async_to_lds_b128 %0, %1, off"
               :: "v"(lo), "v"(g) : "memory");
#endif
}

static __device__ __forceinline__ void cp_wait_all() {
  asm volatile("s_wait_asynccnt 0x0" ::: "memory");
}

// ---------------- WMMA fragment loads (ISA 7.12.2 layouts) ----------------
// A-fragment (16-bit A 16x32): lane L row = L%16; half 0 holds K 0..7 and
// 16..23; half 1 holds K 8..15 and 24..31. base -> [tileRow0][kBase].
static __device__ __forceinline__ v16bf load_a_frag(const unsigned short* base,
                                                    int stride, int lane) {
  const int row = lane & 15, hlf = lane >> 4;
  const unsigned short* p = base + row * stride + hlf * 8;
  Frag f;
  f.q[0] = *(const uint4*)(p);
  f.q[1] = *(const uint4*)(p + 16);
  return f.v;
}

// B-fragment (16-bit B 32x16): lane L col = L%16; half 0 holds K 0..7 and
// 16..23; half 1 holds K 8..15 and 24..31, from transposed tile Bt[N][K].
static __device__ __forceinline__ v16bf load_b_frag(const unsigned short* base,
                                                    int stride, int lane) {
  const int col = lane & 15, hlf = lane >> 4;
  const unsigned short* p = base + col * stride + hlf * 8;
  Frag f;
  f.q[0] = *(const uint4*)(p);
  f.q[1] = *(const uint4*)(p + 16);
  return f.v;
}

static __device__ __forceinline__ v8f wmma_bf16(v16bf a, v16bf b, v8f c) {
  return __builtin_amdgcn_wmma_f32_16x16x32_bf16(false, a, false, b,
                                                 (short)0, c, false, false);
}

// ------------------------------------------------------------------
// weight convert + transpose: W[K][N] fp32 -> Wt[N][K] bf16 (32x32 tiles)
// ------------------------------------------------------------------
__global__ __launch_bounds__(256) void cvt_t_kernel(const float* __restrict__ W,
                                                    unsigned short* __restrict__ Wt,
                                                    int K, int N) {
  __shared__ unsigned short tile[32][33];
  const int n0 = blockIdx.x * 32, k0 = blockIdx.y * 32;
  const int tx = threadIdx.x & 31, ty = threadIdx.x >> 5;
#pragma unroll
  for (int i = 0; i < 4; ++i) {
    int k = ty + i * 8;
    tile[k][tx] = f32_bf16(W[(size_t)(k0 + k) * N + n0 + tx]);
  }
  __syncthreads();
#pragma unroll
  for (int i = 0; i < 4; ++i) {
    int n = ty + i * 8;
    Wt[(size_t)(n0 + n) * K + k0 + tx] = tile[tx][n];
  }
}

// ------------------------------------------------------------------
// V transpose: qkv[b,t, 2C + h*64 + d] -> vt[(b*H+h)*64 + d][t]  (bf16)
// ------------------------------------------------------------------
__global__ __launch_bounds__(256) void vtrans_kernel(const unsigned short* __restrict__ qkv,
                                                     unsigned short* __restrict__ vt) {
  __shared__ unsigned short tile[32][33];
  const int t0 = blockIdx.x * 32, d0 = blockIdx.y * 32;
  const int bh = blockIdx.z;
  const int b = bh / HDIM, h = bh % HDIM;
  const int tx = threadIdx.x & 31, ty = threadIdx.x >> 5;
#pragma unroll
  for (int i = 0; i < 4; ++i) {
    int t = ty + i * 8;
    tile[t][tx] = qkv[((size_t)b * TDIM + t0 + t) * QKVLD + 2 * CDIM + h * DHDIM + d0 + tx];
  }
  __syncthreads();
#pragma unroll
  for (int i = 0; i < 4; ++i) {
    int d = ty + i * 8;
    vt[((size_t)bh * DHDIM + d0 + d) * TDIM + t0 + tx] = tile[tx][d];
  }
}

// ------------------------------------------------------------------
// LayerNorm over rows of length 768 -> bf16
// ------------------------------------------------------------------
__global__ __launch_bounds__(256) void ln_kernel(const float* __restrict__ x,
                                                 const float* __restrict__ w,
                                                 const float* __restrict__ b,
                                                 unsigned short* __restrict__ out) {
  const int row = blockIdx.x;
  const int tid = threadIdx.x;
  const float* xr = x + (size_t)row * CDIM;
  float v0 = xr[tid], v1 = xr[tid + 256], v2 = xr[tid + 512];
  float s = v0 + v1 + v2;
#pragma unroll
  for (int off = 16; off > 0; off >>= 1) s += __shfl_xor(s, off, 32);
  __shared__ float red[8];
  __shared__ float red2[8];
  const int wave = tid >> 5, lane = tid & 31;
  if (lane == 0) red[wave] = s;
  __syncthreads();
  float tot = 0.0f;
#pragma unroll
  for (int i = 0; i < 8; ++i) tot += red[i];
  const float mean = tot * (1.0f / (float)CDIM);
  float d0 = v0 - mean, d1 = v1 - mean, d2 = v2 - mean;
  float q = d0 * d0 + d1 * d1 + d2 * d2;
#pragma unroll
  for (int off = 16; off > 0; off >>= 1) q += __shfl_xor(q, off, 32);
  if (lane == 0) red2[wave] = q;
  __syncthreads();
  float tq = 0.0f;
#pragma unroll
  for (int i = 0; i < 8; ++i) tq += red2[i];
  const float rstd = rsqrtf(tq * (1.0f / (float)CDIM) + 1e-5f);
  unsigned short* orow = out + (size_t)row * CDIM;
  orow[tid]       = f32_bf16(d0 * rstd * w[tid]       + b[tid]);
  orow[tid + 256] = f32_bf16(d1 * rstd * w[tid + 256] + b[tid + 256]);
  orow[tid + 512] = f32_bf16(d2 * rstd * w[tid + 512] + b[tid + 512]);
}

// ------------------------------------------------------------------
// Tiled bf16 GEMM, B pre-transposed: out = epi(A[M,K] @ Bt[N,K]^T + bias)
// Block tile 128x128, BK=32, double-buffered async global->LDS staging.
// 8 waves in 2(M)x4(N) grid; 8 wmma per K-step.
// ------------------------------------------------------------------
template <int ACT, int RESID, int OUTBF>
__global__ __launch_bounds__(256) void gemm_kernel(
    const unsigned short* __restrict__ A, const unsigned short* __restrict__ Bt,
    const float* __restrict__ bias, const float* __restrict__ resid,
    void* __restrict__ out, int M, int N, int K) {
  __shared__ __align__(16) unsigned short As[2][128 * 40];  // [row][k]
  __shared__ __align__(16) unsigned short Bs[2][128 * 40];  // [n][k]

  const int n0 = blockIdx.x * 128;
  const int m0 = blockIdx.y * 128;
  const int wave = threadIdx.x >> 5, lane = threadIdx.x & 31;
  const int wm = wave & 1, wn = wave >> 1;

  v8f acc[4][2];
  v8f zf;
#pragma unroll
  for (int i = 0; i < 8; ++i) zf[i] = 0.0f;
#pragma unroll
  for (int mt = 0; mt < 4; ++mt)
#pragma unroll
    for (int nt = 0; nt < 2; ++nt) acc[mt][nt] = zf;

  auto stage = [&](int k0, int buf) {
#pragma unroll
    for (int i = 0; i < 2; ++i) {
      int c = threadIdx.x + i * 256;             // 0..511 chunks of 8 bf16
      int row = c >> 2, kc = (c & 3) << 3;
      cp_async16(A  + (size_t)(m0 + row) * K + k0 + kc, &As[buf][row * 40 + kc]);
      cp_async16(Bt + (size_t)(n0 + row) * K + k0 + kc, &Bs[buf][row * 40 + kc]);
    }
  };

  stage(0, 0);
  cp_wait_all();
  __syncthreads();

  int cur = 0;
  for (int k0 = 0; k0 < K; k0 += 32) {
    if (k0 + 32 < K) stage(k0 + 32, cur ^ 1);    // overlap with wmma below

    const unsigned short* Abase = As[cur] + (wm * 64) * 40;
    const unsigned short* Bbase = Bs[cur] + (wn * 32) * 40;
    v16bf af[4], bfr[2];
#pragma unroll
    for (int mt = 0; mt < 4; ++mt) af[mt] = load_a_frag(Abase + mt * 16 * 40, 40, lane);
#pragma unroll
    for (int nt = 0; nt < 2; ++nt) bfr[nt] = load_b_frag(Bbase + nt * 16 * 40, 40, lane);
#pragma unroll
    for (int mt = 0; mt < 4; ++mt)
#pragma unroll
      for (int nt = 0; nt < 2; ++nt)
        acc[mt][nt] = wmma_bf16(af[mt], bfr[nt], acc[mt][nt]);

    cp_wait_all();
    __syncthreads();
    cur ^= 1;
  }

  // epilogue: C layout VGPR r <-> row r (lanes 0-15) / r+8 (lanes 16-31)
  const int hlf = lane >> 4, lnn = lane & 15;
#pragma unroll
  for (int mt = 0; mt < 4; ++mt) {
#pragma unroll
    for (int nt = 0; nt < 2; ++nt) {
      const int col = n0 + wn * 32 + nt * 16 + lnn;
      const float bv = bias[col];
#pragma unroll
      for (int r = 0; r < 8; ++r) {
        const int row = m0 + wm * 64 + mt * 16 + r + 8 * hlf;
        float v = acc[mt][nt][r] + bv;
        if (ACT == 1) v = 0.5f * v * (1.0f + erff(v * 0.70710678118654752f));
        if (RESID) v += resid[(size_t)row * N + col];
        if (OUTBF)
          ((unsigned short*)out)[(size_t)row * N + col] = f32_bf16(v);
        else
          ((float*)out)[(size_t)row * N + col] = v;
      }
    }
  }
}

// ------------------------------------------------------------------
// Flash attention: one block per (b, h, 128-row Q tile), 8 waves each
// owning 16 query rows; key blocks of 64; online softmax in registers.
// K/V tiles staged via async global->LDS (V pre-transposed in vt).
// ------------------------------------------------------------------
__global__ __launch_bounds__(256) void attn_kernel(
    const unsigned short* __restrict__ qkv, const unsigned short* __restrict__ vt,
    unsigned short* __restrict__ y) {
  __shared__ __align__(16) unsigned short Qs[128 * 72];  // [qrow][dh]
  __shared__ __align__(16) unsigned short Ks[64 * 72];   // [key][dh]  (= Bt for QK^T)
  __shared__ __align__(16) unsigned short Vts[64 * 72];  // [dh][key]  (= Bt for P@V)
  __shared__ __align__(16) unsigned short Ps[128 * 72];  // [qrow][key]

  const int qblk = blockIdx.x;   // 0..15
  const int hh = blockIdx.y;     // 0..11
  const int bb = blockIdx.z;     // 0..3
  const int bh = bb * HDIM + hh;
  const int wave = threadIdx.x >> 5, lane = threadIdx.x & 31;
  const int hlf = lane >> 4, lnn = lane & 15;
  const size_t qrow0 = (size_t)bb * TDIM + qblk * 128;

  // stage Q tile once (async; first loop iteration's wait covers it)
#pragma unroll
  for (int i = 0; i < 4; ++i) {
    int c = threadIdx.x + i * 256;
    int row = c >> 3, ch = (c & 7) << 3;
    cp_async16(qkv + (qrow0 + row) * QKVLD + hh * DHDIM + ch, &Qs[row * 72 + ch]);
  }

  float m[8], l[8];
  v8f O[4];
  v8f zf;
#pragma unroll
  for (int i = 0; i < 8; ++i) zf[i] = 0.0f;
#pragma unroll
  for (int t = 0; t < 4; ++t) O[t] = zf;
#pragma unroll
  for (int r = 0; r < 8; ++r) { m[r] = -INFINITY; l[r] = 0.0f; }

  for (int kb = 0; kb < TDIM / 64; ++kb) {
    // stage K block [key][dh] and Vt block [dh][key] (both contiguous rows)
#pragma unroll
    for (int i = 0; i < 2; ++i) {
      int c = threadIdx.x + i * 256;
      int r = c >> 3, ch = (c & 7) << 3;
      cp_async16(qkv + ((size_t)bb * TDIM + kb * 64 + r) * QKVLD + CDIM + hh * DHDIM + ch,
                 &Ks[r * 72 + ch]);
      cp_async16(vt + ((size_t)bh * DHDIM + r) * TDIM + kb * 64 + ch,
                 &Vts[r * 72 + ch]);
    }
    cp_wait_all();
    __syncthreads();

    // S[16 rows x 64 keys] (K-dim = Dh = 64 -> 2 wmma steps)
    v8f S[4];
#pragma unroll
    for (int nt = 0; nt < 4; ++nt) S[nt] = zf;
#pragma unroll
    for (int ks = 0; ks < 2; ++ks) {
      v16bf a = load_a_frag(Qs + wave * 16 * 72 + ks * 32, 72, lane);
#pragma unroll
      for (int nt = 0; nt < 4; ++nt) {
        v16bf bfr = load_b_frag(Ks + nt * 16 * 72 + ks * 32, 72, lane);
        S[nt] = wmma_bf16(a, bfr, S[nt]);
      }
    }

    // row max across 4 tiles + the 16 lanes of each half
    float rmax[8];
#pragma unroll
    for (int r = 0; r < 8; ++r) {
      float v = S[0][r];
      v = fmaxf(v, S[1][r]);
      v = fmaxf(v, S[2][r]);
      v = fmaxf(v, S[3][r]);
      rmax[r] = v * 0.125f;                      // 1/sqrt(Dh)
    }
#pragma unroll
    for (int off = 1; off < 16; off <<= 1)
#pragma unroll
      for (int r = 0; r < 8; ++r)
        rmax[r] = fmaxf(rmax[r], __shfl_xor(rmax[r], off, 32));

    float mn[8], sc[8], rsum[8];
#pragma unroll
    for (int r = 0; r < 8; ++r) {
      mn[r] = fmaxf(m[r], rmax[r]);
      sc[r] = __expf(m[r] - mn[r]);
      rsum[r] = 0.0f;
    }
#pragma unroll
    for (int nt = 0; nt < 4; ++nt)
#pragma unroll
      for (int r = 0; r < 8; ++r) {
        float p = __expf(S[nt][r] * 0.125f - mn[r]);
        S[nt][r] = p;
        rsum[r] += p;
      }
#pragma unroll
    for (int off = 1; off < 16; off <<= 1)
#pragma unroll
      for (int r = 0; r < 8; ++r) rsum[r] += __shfl_xor(rsum[r], off, 32);
#pragma unroll
    for (int r = 0; r < 8; ++r) {
      l[r] = l[r] * sc[r] + rsum[r];
      m[r] = mn[r];
    }
#pragma unroll
    for (int t = 0; t < 4; ++t)
#pragma unroll
      for (int r = 0; r < 8; ++r) O[t][r] *= sc[r];

    // spill P (C layout) -> LDS so it can be re-read in A layout (wave-local)
#pragma unroll
    for (int nt = 0; nt < 4; ++nt)
#pragma unroll
      for (int r = 0; r < 8; ++r)
        Ps[(wave * 16 + r + 8 * hlf) * 72 + nt * 16 + lnn] = f32_bf16(S[nt][r]);

    // O += P @ V (K-dim = 64 keys -> 2 wmma steps)
#pragma unroll
    for (int ks = 0; ks < 2; ++ks) {
      v16bf a = load_a_frag(Ps + wave * 16 * 72 + ks * 32, 72, lane);
#pragma unroll
      for (int t = 0; t < 4; ++t) {
        v16bf bfr = load_b_frag(Vts + t * 16 * 72 + ks * 32, 72, lane);
        O[t] = wmma_bf16(a, bfr, O[t]);
      }
    }
    __syncthreads();   // all waves done with Ks/Vts before next restage
  }

  // normalize and scatter into y[B,T,C] as bf16
#pragma unroll
  for (int t = 0; t < 4; ++t) {
#pragma unroll
    for (int r = 0; r < 8; ++r) {
      const size_t row = qrow0 + wave * 16 + r + 8 * hlf;
      const int col = hh * DHDIM + t * 16 + lnn;
      y[row * CDIM + col] = f32_bf16(O[t][r] / l[r]);
    }
  }
}

// ------------------------------------------------------------------
// launcher
// ------------------------------------------------------------------
extern "C" void kernel_launch(void* const* d_in, const int* in_sizes, int n_in,
                              void* d_out, int out_size, void* d_ws, size_t ws_size,
                              hipStream_t stream) {
  (void)in_sizes; (void)n_in; (void)out_size; (void)ws_size;
  const float* x       = (const float*)d_in[0];
  const float* ln1_w   = (const float*)d_in[1];
  const float* ln1_b   = (const float*)d_in[2];
  const float* W_qkv   = (const float*)d_in[3];
  const float* b_qkv   = (const float*)d_in[4];
  const float* W_attn  = (const float*)d_in[5];
  const float* b_attn  = (const float*)d_in[6];
  const float* ln2_w   = (const float*)d_in[7];
  const float* ln2_b   = (const float*)d_in[8];
  const float* W_fc    = (const float*)d_in[9];
  const float* b_fc    = (const float*)d_in[10];
  const float* W_mlp   = (const float*)d_in[11];
  const float* b_mlp   = (const float*)d_in[12];
  float* out = (float*)d_out;

  char* ws = (char*)d_ws;
  size_t off = 0;
  auto alloc = [&](size_t bytes) {
    size_t r = off;
    off = (off + bytes + 255) & ~(size_t)255;
    return r;
  };
  const size_t o_wqkv = alloc((size_t)CDIM * QKVLD * 2);   // transposed [N][K]
  const size_t o_wat  = alloc((size_t)CDIM * CDIM * 2);
  const size_t o_wfc  = alloc((size_t)CDIM * FFDIM * 2);
  const size_t o_wml  = alloc((size_t)FFDIM * CDIM * 2);
  const size_t o_h    = alloc((size_t)MROWS * CDIM * 2);   // LN1 out, later LN2 out
  const size_t o_qkv  = alloc((size_t)MROWS * QKVLD * 2);  // later start of fc act
  const size_t o_y    = alloc((size_t)MROWS * CDIM * 2);   // fc act spills into here
  const size_t o_vt   = alloc((size_t)MROWS * CDIM * 2);   // V transposed [b,h,dh,t]
  const size_t o_x1   = alloc((size_t)MROWS * CDIM * 4);
  const size_t o_fc   = o_qkv;  // [MROWS, FFDIM] bf16 == qkv region + y region

  unsigned short* wqkvT = (unsigned short*)(ws + o_wqkv);
  unsigned short* watT  = (unsigned short*)(ws + o_wat);
  unsigned short* wfcT  = (unsigned short*)(ws + o_wfc);
  unsigned short* wmlT  = (unsigned short*)(ws + o_wml);
  unsigned short* hbuf  = (unsigned short*)(ws + o_h);
  unsigned short* qkvbuf= (unsigned short*)(ws + o_qkv);
  unsigned short* ybuf  = (unsigned short*)(ws + o_y);
  unsigned short* vtbuf = (unsigned short*)(ws + o_vt);
  unsigned short* fcbuf = (unsigned short*)(ws + o_fc);
  float*          x1buf = (float*)(ws + o_x1);

  // 1) weights: fp32 -> bf16, transposed to [N][K]
  cvt_t_kernel<<<dim3(QKVLD / 32, CDIM / 32), 256, 0, stream>>>(W_qkv, wqkvT, CDIM, QKVLD);
  cvt_t_kernel<<<dim3(CDIM / 32, CDIM / 32), 256, 0, stream>>>(W_attn, watT, CDIM, CDIM);
  cvt_t_kernel<<<dim3(FFDIM / 32, CDIM / 32), 256, 0, stream>>>(W_fc, wfcT, CDIM, FFDIM);
  cvt_t_kernel<<<dim3(CDIM / 32, FFDIM / 32), 256, 0, stream>>>(W_mlp, wmlT, FFDIM, CDIM);

  // 2) LN1
  ln_kernel<<<MROWS, 256, 0, stream>>>(x, ln1_w, ln1_b, hbuf);

  // 3) QKV GEMM: [8192,768] x [768,2304] -> bf16
  gemm_kernel<0, 0, 1><<<dim3(QKVLD / 128, MROWS / 128), 256, 0, stream>>>(
      hbuf, wqkvT, b_qkv, nullptr, qkvbuf, MROWS, QKVLD, CDIM);

  // 3b) transpose V per (b,h): vt[bh][dh][t]
  vtrans_kernel<<<dim3(TDIM / 32, DHDIM / 32, BDIM * HDIM), 256, 0, stream>>>(qkvbuf, vtbuf);

  // 4) flash attention -> y bf16 [8192,768]
  attn_kernel<<<dim3(TDIM / 128, HDIM, BDIM), 256, 0, stream>>>(qkvbuf, vtbuf, ybuf);

  // 5) attn proj + residual: x1 = x + y @ W_attn + b (fp32 out)
  gemm_kernel<0, 1, 0><<<dim3(CDIM / 128, MROWS / 128), 256, 0, stream>>>(
      ybuf, watT, b_attn, x, x1buf, MROWS, CDIM, CDIM);

  // 6) LN2
  ln_kernel<<<MROWS, 256, 0, stream>>>(x1buf, ln2_w, ln2_b, hbuf);

  // 7) FC + exact GELU: [8192,768] x [768,3072] -> bf16
  gemm_kernel<1, 0, 1><<<dim3(FFDIM / 128, MROWS / 128), 256, 0, stream>>>(
      hbuf, wfcT, b_fc, nullptr, fcbuf, MROWS, FFDIM, CDIM);

  // 8) MLP proj + residual -> d_out fp32
  gemm_kernel<0, 1, 0><<<dim3(CDIM / 128, MROWS / 128), 256, 0, stream>>>(
      fcbuf, wmlT, b_mlp, x1buf, out, MROWS, CDIM, FFDIM);
}
